// NeuralCDE_43757126812122
// MI455X (gfx1250) — matrix operand order repaired and checked by reading, not compiled, not run on hardware
//
#include <hip/hip_runtime.h>

// ---------------------------------------------------------------------------
// NeuralCDE scan for MI455X (gfx1250, wave32, WMMA)
//   B=256, T=256, D=16, H=128.
//   16 workgroups x 256 threads; each WG owns 16 batch rows and runs the full
//   255-step scan with z resident in LDS. GEMMs use v_wmma_f32_16x16x32_f16
//   with W1/W2 pre-packed (f32->f16) into B-fragment lane order in workspace.
// ---------------------------------------------------------------------------

typedef _Float16 v16h __attribute__((ext_vector_type(16)));
typedef _Float16 f16x4 __attribute__((ext_vector_type(4)));
typedef float    v8f  __attribute__((ext_vector_type(8)));
typedef float    f32x4 __attribute__((ext_vector_type(4)));

#define BB   256
#define TT   256
#define DD   16
#define HH   128
#define H2   256            // 2H
#define ND   2048           // H*D
#define MBLK 16             // batch rows per workgroup
#define NWAVE 8
#define ZPAD 132            // H + 4 floats  (row stride 528B -> bank-conflict free)
#define HPAD 264            // H2 + 8 halves (row stride 528B)

#define KC1 (HH/32)         // 4 k-chunks for GEMM1
#define KC2 (H2/32)         // 8 k-chunks for GEMM2
#define NT2 (ND/16)         // 128 n-tiles for GEMM2

#if __has_builtin(__builtin_amdgcn_tanhf)
#define FAST_TANH(x) __builtin_amdgcn_tanhf(x)     // hardware V_TANH_F32
#else
#define FAST_TANH(x) tanhf(x)
#endif

union FragU {
    v16h  v;
    f16x4 h4[4];
    f32x4 q[2];   // raw 16B copy container
};

// Pack row-major f32 (K x ncols) weight into f16 WMMA B-fragments:
// dst[((nt*kchunks + kc)*32 + lane)*16 + e] = W[(32*kc + (lane>=16?8:0) + e + (e>=8?8:0))*ncols + 16*nt + (lane&15)]
__global__ __launch_bounds__(256)
void pack_bfrag_kernel(const float* __restrict__ src, _Float16* __restrict__ dst,
                       int kchunks, int ntiles, int ncols)
{
    int idx = blockIdx.x * 256 + threadIdx.x;
    int total = ntiles * kchunks * 32 * 16;
    if (idx >= total) return;
    int e  = idx & 15;
    int l  = (idx >> 4) & 31;
    int kc = (idx >> 9) % kchunks;
    int nt = (idx >> 9) / kchunks;
    int K  = 32 * kc + ((l >= 16) ? 8 : 0) + e + ((e >= 8) ? 8 : 0);
    int n  = 16 * nt + (l & 15);
    dst[idx] = (_Float16)src[(size_t)K * ncols + n];
}

__global__ __launch_bounds__(256)
void cde_scan_kernel(const float* __restrict__ x,
                     const float* __restrict__ W_enc,
                     const float* __restrict__ b_enc,
                     const float* __restrict__ b1,
                     const float* __restrict__ b2,
                     const _Float16* __restrict__ w1f,
                     const _Float16* __restrict__ w2f,
                     float* __restrict__ out)
{
    __shared__ __align__(16) float    z_lds[MBLK * ZPAD];   // f32 state
    __shared__ __align__(16) _Float16 h_lds[MBLK * HPAD];   // f16 hidden (A of GEMM2)
    __shared__ __align__(16) float    dx_lds[MBLK * DD];
    __shared__ __align__(16) float    b1_lds[H2];
    __shared__ __align__(16) float    b2_lds[ND];

    const int tid     = threadIdx.x;
    const int lane    = tid & 31;
    const int wv      = tid >> 5;          // wave 0..7
    const int halfsel = lane >> 4;         // 0: lanes 0-15, 1: lanes 16-31
    const int ln      = lane & 15;
    const int b0      = blockIdx.x * MBLK;

    // biases -> LDS
    if (tid < H2) b1_lds[tid] = b1[tid];
    for (int i = tid; i < ND; i += 256) b2_lds[i] = b2[i];

    // per-thread dx element coordinates (m = tid>>4, d = tid&15)
    const float* x_dx = x + (size_t)(b0 + (tid >> 4)) * TT * DD + (tid & 15);
    const int    dxslot = (tid >> 4) * DD + (tid & 15);

    // ---- z0 = softplus(x[:,0,:] @ W_enc + b_enc) : 16x128, scalar f32 ----
    #pragma unroll
    for (int i = 0; i < (MBLK * HH) / 256; ++i) {
        int idx = tid + 256 * i;
        int m = idx >> 7;           // /128
        int c = idx & (HH - 1);
        const float* xr = x + (size_t)(b0 + m) * TT * DD;   // t = 0
        float acc = b_enc[c];
        #pragma unroll
        for (int d = 0; d < DD; ++d) acc += xr[d] * W_enc[d * HH + c];
        // numerically stable softplus
        float sp = fmaxf(acc, 0.f) + log1pf(__expf(-fabsf(acc)));
        z_lds[m * ZPAD + c] = sp;
    }

    // ---- scan over T-1 steps ----
    for (int t = 0; t < TT - 1; ++t) {
        // issue dx global loads BEFORE the barrier (latency hides under it);
        // the LDS store happens after the barrier.
        float xnext = x_dx[(t + 1) * DD];
        float xcur  = x_dx[t * DD];

        __syncthreads();                       // z from previous step is published

        dx_lds[dxslot] = xnext - xcur;

        // ---- GEMM1: h = silu(z @ W1 + b1)  (16x128)@(128x256) ----
        FragU a1[KC1];
        {
            const float* zr = z_lds + ln * ZPAD;          // A: lane -> row M=ln
            #pragma unroll
            for (int c = 0; c < KC1; ++c) {
                int kb = 32 * c + halfsel * 8;
                a1[c].h4[0] = __builtin_convertvector(*(const f32x4*)(zr + kb),      f16x4);
                a1[c].h4[1] = __builtin_convertvector(*(const f32x4*)(zr + kb + 4),  f16x4);
                a1[c].h4[2] = __builtin_convertvector(*(const f32x4*)(zr + kb + 16), f16x4);
                a1[c].h4[3] = __builtin_convertvector(*(const f32x4*)(zr + kb + 20), f16x4);
            }
        }
        #pragma unroll
        for (int it = 0; it < 2; ++it) {                  // 2 n-tiles per wave
            int nt = wv * 2 + it;
            v8f acc = {};
            const _Float16* wt = w1f + (size_t)nt * KC1 * 512;
            #pragma unroll
            for (int c = 0; c < KC1; ++c) {
                FragU bf;
                const f32x4* p = (const f32x4*)(wt + c * 512 + lane * 16);
                bf.q[0] = p[0]; bf.q[1] = p[1];
                acc = __builtin_amdgcn_wmma_f32_16x16x32_f16(
                        false, a1[c].v, false, bf.v, (short)0, acc, false, false);
            }
            float bv = b1_lds[nt * 16 + ln];
            #pragma unroll
            for (int r = 0; r < 8; ++r) {
                int m = r + 8 * halfsel;
                float v = acc[r] + bv;
                float s = v * __builtin_amdgcn_rcpf(1.f + __expf(-v));   // silu
                h_lds[m * HPAD + nt * 16 + ln] = (_Float16)s;
            }
        }

        __syncthreads();                       // h and dx published; z reads done

        // dx in C-matrix lane layout (shared across all 16 tiles)
        float dxc[8];
        #pragma unroll
        for (int r = 0; r < 8; ++r) dxc[r] = dx_lds[(r + 8 * halfsel) * DD + ln];

        // A-fragments of h for GEMM2 (held in registers across all n-tiles)
        FragU a2[KC2];
        {
            const _Float16* hr = h_lds + ln * HPAD;
            #pragma unroll
            for (int c = 0; c < KC2; ++c) {
                int kb = 32 * c + halfsel * 8;
                a2[c].q[0] = *(const f32x4*)(hr + kb);
                a2[c].q[1] = *(const f32x4*)(hr + kb + 16);
            }
        }

        // ---- GEMM2 + tanh + dz contraction; tile j == hidden index j ----
        #pragma unroll 1
        for (int i = 0; i < NT2 / NWAVE; ++i) {
            int j = i * NWAVE + wv;                        // waves march adjacently in L2
            v8f acc = {};
            const _Float16* wt = w2f + (size_t)j * KC2 * 512;
            #pragma unroll
            for (int c = 0; c < KC2; ++c) {
                FragU bf;
                const f32x4* p = (const f32x4*)(wt + c * 512 + lane * 16);
                bf.q[0] = p[0]; bf.q[1] = p[1];
                acc = __builtin_amdgcn_wmma_f32_16x16x32_f16(
                        false, a2[c].v, false, bf.v, (short)0, acc, false, false);
            }
            float bv = b2_lds[j * 16 + ln];
            #pragma unroll
            for (int r = 0; r < 8; ++r) {
                float mval = FAST_TANH(acc[r] + bv);       // mat[b, j, d]
                float p = mval * dxc[r];                   // * dx[b, d]
                p += __shfl_xor(p, 1, 32);                 // reduce over d (16 lanes)
                p += __shfl_xor(p, 2, 32);
                p += __shfl_xor(p, 4, 32);
                p += __shfl_xor(p, 8, 32);
                if (ln == 0) {
                    int m = r + 8 * halfsel;
                    z_lds[m * ZPAD + j] += p;              // unique (m, j) owner
                }
            }
        }
    }

    __syncthreads();
    #pragma unroll
    for (int i = 0; i < (MBLK * HH) / 256; ++i) {
        int idx = tid + 256 * i;
        int m = idx >> 7, c = idx & (HH - 1);
        out[(size_t)(b0 + m) * HH + c] = z_lds[m * ZPAD + c];
    }
}

extern "C" void kernel_launch(void* const* d_in, const int* in_sizes, int n_in,
                              void* d_out, int out_size, void* d_ws, size_t ws_size,
                              hipStream_t stream)
{
    (void)in_sizes; (void)n_in; (void)out_size; (void)ws_size;
    const float* x     = (const float*)d_in[0];
    const float* W_enc = (const float*)d_in[1];
    const float* b_enc = (const float*)d_in[2];
    const float* W1    = (const float*)d_in[3];
    const float* b1    = (const float*)d_in[4];
    const float* W2    = (const float*)d_in[5];
    const float* b2    = (const float*)d_in[6];
    float* out = (float*)d_out;

    // workspace: [0, 1MB) W2 fragments, [1MB, 1MB+64KB) W1 fragments (f16)
    const size_t w2_elems = (size_t)NT2 * KC2 * 32 * 16;   // 524288 halves = 1 MB
    const size_t w1_elems = (size_t)16  * KC1 * 32 * 16;   // 32768 halves  = 64 KB
    _Float16* w2f = (_Float16*)d_ws;
    _Float16* w1f = (_Float16*)((char*)d_ws + w2_elems * sizeof(_Float16));

    pack_bfrag_kernel<<<(int)(w2_elems / 256), 256, 0, stream>>>(W2, w2f, KC2, NT2, ND);
    pack_bfrag_kernel<<<(int)(w1_elems / 256), 256, 0, stream>>>(W1, w1f, KC1, 16, H2);
    cde_scan_kernel<<<BB / MBLK, 256, 0, stream>>>(x, W_enc, b_enc, b1, b2, w1f, w2f, out);
}
